// OctoSS2D_8392366097074
// MI455X (gfx1250) — matrix operand order repaired and checked by reading, not compiled, not run on hardware
//
#include <hip/hip_runtime.h>

// ---------------- problem constants ----------------
#define BATCH   8
#define CMODEL  192
#define HH      64
#define WW      64
#define LSEQ    4096            // HH*WW
#define NTOK    (BATCH*LSEQ)    // 32768
#define DINNER  384
#define DSTATE  16
#define DTRANK  12
#define NXPROJ  44              // DTRANK + 2*DSTATE

typedef _Float16 v16h __attribute__((ext_vector_type(16)));
typedef _Float16 v8h  __attribute__((ext_vector_type(8)));
typedef float    v8f  __attribute__((ext_vector_type(8)));

// fast sigmoid / silu: single v_rcp_f32 instead of IEEE divide sequence
__device__ __forceinline__ float fast_sigmoid(float v) {
    return __builtin_amdgcn_rcpf(1.0f + __expf(-v));
}
__device__ __forceinline__ float fast_silu(float v) {
    return v * fast_sigmoid(v);
}

// token index l -> pixel (r, cc) bijection for direction dir (same map for
// gather and scatter). dir = 2*pat + flip, pat: 0=h 1=v 2=tlbr 3=trbl.
__device__ __forceinline__ void octo_pix(int l, int dir, int& r, int& cc) {
    int lf = (dir & 1) ? (LSEQ - 1 - l) : l;
    int pat = dir >> 1;
    if (pat == 1) { r = lf & 63; cc = lf >> 6; return; }
    r = lf >> 6;
    int col = lf & 63;
    if      (pat == 0) cc = col;
    else if (pat == 2) cc = (col - r) & 63;   // (col - r) mod 64
    else               cc = (col + r) & 63;   // (col + r) mod 64
}

// ---------------- zero output ----------------
__global__ void k_zero(float* __restrict__ p, int n) {
    int i = blockIdx.x * blockDim.x + threadIdx.x;
    if (i < n) p[i] = 0.0f;
}

// ---------------- weight conversion to f16 (row-major [N][K]) ----------------
__global__ void k_cvt_w(const float* __restrict__ w_in, const float* __restrict__ w_out,
                        _Float16* __restrict__ wA, _Float16* __restrict__ wO) {
    int i = blockIdx.x * blockDim.x + threadIdx.x;
    const int n1 = 2 * DINNER * CMODEL;   // 768*192
    const int n2 = CMODEL * DINNER;       // 192*384
    if (i < n1)            wA[i]      = (_Float16)w_in[i];
    else if (i < n1 + n2)  wO[i - n1] = (_Float16)w_out[i - n1];
}

// ---------------- gather + layernorm -> f16 sequence ----------------
__global__ __launch_bounds__(192) void k_gather_ln(
    const float* __restrict__ x, const float* __restrict__ ln_w,
    const float* __restrict__ ln_b, _Float16* __restrict__ seq, int dir) {
    __shared__ float s1[256], s2[256];
    int tok = blockIdx.x;
    int b = tok >> 12, l = tok & (LSEQ - 1);
    int r, cc; octo_pix(l, dir, r, cc);
    int c = threadIdx.x;
    float v = x[(((size_t)b * CMODEL + c) << 12) + (r << 6) + cc];
    s1[c] = v; s2[c] = v * v;
    if (c < 64) { s1[192 + c] = 0.0f; s2[192 + c] = 0.0f; }
    __syncthreads();
    for (int st = 128; st > 0; st >>= 1) {
        if (c < st) { s1[c] += s1[c + st]; s2[c] += s2[c + st]; }
        __syncthreads();
    }
    float mean = s1[0] * (1.0f / CMODEL);
    float var  = s2[0] * (1.0f / CMODEL) - mean * mean;
    float nv = (v - mean) * rsqrtf(var + 1e-5f) * ln_w[c] + ln_b[c];
    seq[(size_t)tok * CMODEL + c] = (_Float16)nv;
}

// ---------------- in_proj: [NTOK,192]f16 x [192,768] -> xc (f32), silu(z) (f32)
// one wave = one 16x16 D tile, WMMA f32_16x16x32_f16, K = 192 = 6 steps.
__global__ __launch_bounds__(256) void k_inproj(
    const _Float16* __restrict__ seq, const _Float16* __restrict__ wA,
    float* __restrict__ xc_raw, float* __restrict__ sz) {
    int wave = threadIdx.x >> 5;
    int lane = threadIdx.x & 31;
    int n_base = blockIdx.x * 128 + wave * 16;   // N = 768
    int m_base = blockIdx.y * 16;                // M = 32768
    int hi = lane >> 4, lo = lane & 15;
    v8f acc;
    #pragma unroll
    for (int i = 0; i < 8; ++i) acc[i] = 0.0f;
    const _Float16* arow = seq + (size_t)(m_base + lo) * CMODEL;
    const _Float16* brow = wA  + (size_t)(n_base + lo) * CMODEL;
    #pragma unroll
    for (int kk = 0; kk < CMODEL; kk += 32) {
        v8h a0 = *(const v8h*)(arow + kk + hi * 16);       // K = kk + hi*16 .. +7
        v8h a1 = *(const v8h*)(arow + kk + hi * 16 + 8);   // K = kk+hi*16+8 .. +15
        v16h a;
        #pragma unroll
        for (int j = 0; j < 8; ++j) { a[j] = a0[j]; a[8 + j] = a1[j]; }
        v16h bf = *(const v16h*)(brow + kk + hi * 16);     // B[k][n] = W[n][k]
        acc = __builtin_amdgcn_wmma_f32_16x16x32_f16(false, a, false, bf,
                                                     (short)0, acc, false, false);
    }
    int n = n_base + lo;
    #pragma unroll
    for (int i = 0; i < 8; ++i) {
        size_t tok = (size_t)m_base + i + 8 * hi;
        float v = acc[i];
        if (n < DINNER) {
            xc_raw[tok * DINNER + n] = v;
        } else {
            sz[tok * DINNER + (n - DINNER)] = fast_silu(v);  // silu(z)
        }
    }
}

// ---------------- conv(k=4,causal)+silu, x_proj, dt_proj+softplus ----------------
#define CT 8
__global__ __launch_bounds__(256) void k_conv_xproj(
    const float* __restrict__ xc_raw, const float* __restrict__ conv_w,
    const float* __restrict__ conv_b, const float* __restrict__ x_proj_w,
    const float* __restrict__ dt_proj_w, const float* __restrict__ dt_proj_b,
    float* __restrict__ xs, float* __restrict__ Bm, float* __restrict__ Cm,
    float* __restrict__ dtg) {
    __shared__ float xcs[(CT + 3) * DINNER];
    __shared__ float xss[CT * DINNER];
    __shared__ float wcv[DINNER * 4];
    __shared__ float cb[DINNER];
    __shared__ float dtl[CT * DTRANK];
    int tid = threadIdx.x;
    int t0 = blockIdx.x * CT;                 // blocks never cross batch (4096 % 8 == 0)
    int bstart = (t0 >> 12) << 12;
    for (int i = tid; i < DINNER * 4; i += 256) wcv[i] = conv_w[i];
    for (int i = tid; i < DINNER;     i += 256) cb[i]  = conv_b[i];
    for (int i = tid; i < (CT + 3) * DINNER; i += 256) {
        int tt = t0 - 3 + i / DINNER, ch = i % DINNER;
        xcs[i] = (tt >= bstart) ? xc_raw[(size_t)tt * DINNER + ch] : 0.0f;
    }
    __syncthreads();
    for (int i = tid; i < CT * DINNER; i += 256) {
        int t = i / DINNER, ch = i % DINNER;
        float s = cb[ch];
        #pragma unroll
        for (int j = 0; j < 4; ++j) s += wcv[ch * 4 + j] * xcs[(t + j) * DINNER + ch];
        s = fast_silu(s);
        xss[i] = s;
        xs[(size_t)(t0 + t) * DINNER + ch] = s;
    }
    __syncthreads();
    for (int o = tid; o < CT * NXPROJ; o += 256) {
        int t = o / NXPROJ, j = o % NXPROJ;
        const float* wrow = x_proj_w + j * DINNER;
        const float* xr = xss + t * DINNER;
        float acc = 0.0f;
        for (int k = 0; k < DINNER; ++k) acc += wrow[k] * xr[k];
        if (j < DTRANK)               dtl[t * DTRANK + j] = acc;
        else if (j < DTRANK + DSTATE) Bm[(size_t)(t0 + t) * DSTATE + (j - DTRANK)] = acc;
        else                          Cm[(size_t)(t0 + t) * DSTATE + (j - DTRANK - DSTATE)] = acc;
    }
    __syncthreads();
    for (int o = tid; o < CT * DINNER; o += 256) {
        int t = o / DINNER, d = o % DINNER;
        float acc = dt_proj_b[d];
        #pragma unroll
        for (int r2 = 0; r2 < DTRANK; ++r2) acc += dt_proj_w[d * DTRANK + r2] * dtl[t * DTRANK + r2];
        float sp = (acc > 20.0f) ? acc : log1pf(__expf(acc));   // softplus
        dtg[(size_t)(t0 + t) * DINNER + d] = sp;
    }
}

// ---------------- selective scan + gating (fused, state-parallel) ----------------
// 8 lanes per channel, 2 states per lane; per-step y reduced with wave shuffles
// (no barriers). sub==0 lane applies D-skip + silu(z) gate and stores f16.
__global__ __launch_bounds__(256) void k_scan(
    const float* __restrict__ dtg, const float* __restrict__ xs,
    const float* __restrict__ Bm, const float* __restrict__ Cm,
    const float* __restrict__ A_log, const float* __restrict__ D_skip,
    const float* __restrict__ sz, _Float16* __restrict__ y16) {
    int tid = threadIdx.x;
    int sub = tid & 7;                 // state sub-group: states 2*sub, 2*sub+1
    int d = blockIdx.x * 32 + (tid >> 3);
    int b = blockIdx.y;
    int s0 = sub * 2;
    float a0 = -__expf(A_log[d * DSTATE + s0]);
    float a1 = -__expf(A_log[d * DSTATE + s0 + 1]);
    float h0 = 0.0f, h1 = 0.0f;
    float Dk = D_skip[d];
    size_t base = (size_t)b * LSEQ;
    for (int l = 0; l < LSEQ; ++l) {
        size_t tok = base + l;
        float dt = 0.0f, xv = 0.0f;
        if (sub == 0) {
            dt = dtg[tok * DINNER + d];
            xv = xs[tok * DINNER + d];
        }
        dt = __shfl(dt, 0, 8);         // broadcast within 8-lane state group
        xv = __shfl(xv, 0, 8);
        float dx = dt * xv;
        float2 Bp = *(const float2*)(Bm + tok * DSTATE + s0);
        float2 Cp = *(const float2*)(Cm + tok * DSTATE + s0);
        h0 = __expf(dt * a0) * h0 + dx * Bp.x;
        h1 = __expf(dt * a1) * h1 + dx * Bp.y;
        float y = h0 * Cp.x + h1 * Cp.y;
        y += __shfl_xor(y, 1, 8);
        y += __shfl_xor(y, 2, 8);
        y += __shfl_xor(y, 4, 8);
        if (sub == 0) {
            float yg = (y + xv * Dk) * sz[tok * DINNER + d];
            y16[tok * DINNER + d] = (_Float16)yg;
        }
    }
}

// ---------------- out_proj WMMA + inverse-permute scatter-add ----------------
__global__ __launch_bounds__(128) void k_outproj(
    const _Float16* __restrict__ y16, const _Float16* __restrict__ wO,
    float* __restrict__ out, int dir) {
    int wave = threadIdx.x >> 5;
    int lane = threadIdx.x & 31;
    int n_base = blockIdx.x * 64 + wave * 16;   // N = 192
    int m_base = blockIdx.y * 16;               // M = 32768
    int hi = lane >> 4, lo = lane & 15;
    v8f acc;
    #pragma unroll
    for (int i = 0; i < 8; ++i) acc[i] = 0.0f;
    const _Float16* arow = y16 + (size_t)(m_base + lo) * DINNER;
    const _Float16* brow = wO  + (size_t)(n_base + lo) * DINNER;
    #pragma unroll
    for (int kk = 0; kk < DINNER; kk += 32) {
        v8h a0 = *(const v8h*)(arow + kk + hi * 16);
        v8h a1 = *(const v8h*)(arow + kk + hi * 16 + 8);
        v16h a;
        #pragma unroll
        for (int j = 0; j < 8; ++j) { a[j] = a0[j]; a[8 + j] = a1[j]; }
        v16h bf = *(const v16h*)(brow + kk + hi * 16);
        acc = __builtin_amdgcn_wmma_f32_16x16x32_f16(false, a, false, bf,
                                                     (short)0, acc, false, false);
    }
    int n = n_base + lo;
    #pragma unroll
    for (int i = 0; i < 8; ++i) {
        int tok = m_base + i + 8 * hi;
        int b = tok >> 12, l = tok & (LSEQ - 1);
        int r, cc; octo_pix(l, dir, r, cc);
        // directions are stream-serialized; each (pixel,channel) written once here
        out[(((size_t)b * CMODEL + n) << 12) + (r << 6) + cc] += 0.125f * acc[i];
    }
}

// ---------------- host launcher ----------------
extern "C" void kernel_launch(void* const* d_in, const int* in_sizes, int n_in,
                              void* d_out, int out_size, void* d_ws, size_t ws_size,
                              hipStream_t stream) {
    (void)in_sizes; (void)n_in; (void)out_size; (void)ws_size;
    const float* x          = (const float*)d_in[0];
    const float* ln_w       = (const float*)d_in[1];
    const float* ln_b       = (const float*)d_in[2];
    const float* in_proj_w  = (const float*)d_in[3];
    const float* conv_w     = (const float*)d_in[4];
    const float* conv_b     = (const float*)d_in[5];
    const float* x_proj_w   = (const float*)d_in[6];
    const float* dt_proj_w  = (const float*)d_in[7];
    const float* dt_proj_b  = (const float*)d_in[8];
    const float* A_log      = (const float*)d_in[9];
    const float* D_skip     = (const float*)d_in[10];
    const float* out_proj_w = (const float*)d_in[11];
    float* out = (float*)d_out;

    char* ws = (char*)d_ws;
    size_t off = 0;
    auto carve = [&](size_t bytes) -> void* {
        void* p = ws + off;
        off += (bytes + 255) & ~(size_t)255;
        return p;
    };
    _Float16* wA16   = (_Float16*)carve((size_t)2 * DINNER * CMODEL * 2); // 768x192 f16
    _Float16* wO16   = (_Float16*)carve((size_t)CMODEL * DINNER * 2);     // 192x384 f16
    _Float16* seq16  = (_Float16*)carve((size_t)NTOK * CMODEL * 2);
    float*    xc_raw = (float*)   carve((size_t)NTOK * DINNER * 4);
    float*    xs     = (float*)   carve((size_t)NTOK * DINNER * 4);
    float*    sz     = (float*)   carve((size_t)NTOK * DINNER * 4);
    float*    dtg    = (float*)   carve((size_t)NTOK * DINNER * 4);
    float*    Bm     = (float*)   carve((size_t)NTOK * DSTATE * 4);
    float*    Cm     = (float*)   carve((size_t)NTOK * DSTATE * 4);
    _Float16* y16    = (_Float16*)carve((size_t)NTOK * DINNER * 2);

    int nout = BATCH * CMODEL * HH * WW;
    k_zero<<<(nout + 255) / 256, 256, 0, stream>>>(out, nout);
    int ncvt = 2 * DINNER * CMODEL + CMODEL * DINNER;
    k_cvt_w<<<(ncvt + 255) / 256, 256, 0, stream>>>(in_proj_w, out_proj_w, wA16, wO16);

    for (int dir = 0; dir < 8; ++dir) {
        k_gather_ln<<<NTOK, 192, 0, stream>>>(x, ln_w, ln_b, seq16, dir);
        k_inproj<<<dim3(6, NTOK / 16), 256, 0, stream>>>(seq16, wA16, xc_raw, sz);
        k_conv_xproj<<<NTOK / CT, 256, 0, stream>>>(xc_raw, conv_w, conv_b, x_proj_w,
                                                    dt_proj_w, dt_proj_b, xs, Bm, Cm, dtg);
        k_scan<<<dim3(DINNER / 32, BATCH), 256, 0, stream>>>(dtg, xs, Bm, Cm,
                                                             A_log, D_skip, sz, y16);
        k_outproj<<<dim3(3, NTOK / 16), 128, 0, stream>>>(y16, wO16, out, dir);
    }
}
